// KPCN_59768764891745
// MI455X (gfx1250) — matrix operand order repaired and checked by reading, not compiled, not run on hardware
//
#include <hip/hip_runtime.h>
#include <hip/hip_bf16.h>
#include <stdint.h>

typedef __attribute__((ext_vector_type(16))) _Float16 v16h;
typedef __attribute__((ext_vector_type(8)))  _Float16 v8h;
typedef __attribute__((ext_vector_type(8)))  float    v8f;

static constexpr int kHW   = 192;   // image H == W
static constexpr int kCPAD = 128;   // channel pad (100 -> 128)
static constexpr int kCSTR = 136;   // LDS per-pixel stride (bank spread)

static __device__ __forceinline__ v16h cat8(v8h lo, v8h hi) {
    return __builtin_shufflevector(lo, hi, 0,1,2,3,4,5,6,7,8,9,10,11,12,13,14,15);
}
// A-fragment: 8 halves at p0, 8 halves at p1 (K and K+8 ranges)
static __device__ __forceinline__ v16h ldA(const _Float16* p0, const _Float16* p1) {
    return cat8(*(const v8h*)p0, *(const v8h*)p1);
}
// B-fragment: 16 contiguous halves (K..K+15)
static __device__ __forceinline__ v16h ldB(const _Float16* p) {
    return cat8(*(const v8h*)p, *(const v8h*)(p + 8));
}
static __device__ __forceinline__ v8f wmma16(v16h a, v16h b, v8f c) {
    return __builtin_amdgcn_wmma_f32_16x16x32_f16(false, a, false, b, (short)0, c,
                                                  false, false);
}

// ---- weight conversion: w_hidden f32 [7][100][100][5][5] -> f16 [7][25][128][128]
__global__ void cvt_whid_k(const float* __restrict__ w, _Float16* __restrict__ o) {
    size_t i = (size_t)blockIdx.x * blockDim.x + threadIdx.x;
    const size_t total = (size_t)7 * 25 * 128 * 128;
    if (i >= total) return;
    int ic = i & 127;
    int oc = (int)((i >> 7) & 127);
    int t  = (int)((i >> 14) % 25);
    int l  = (int)(i / ((size_t)25 * 128 * 128));
    float v = 0.f;
    if (ic < 100 && oc < 100) {
        int ky = t / 5, kx = t % 5;
        v = w[((((size_t)l * 100 + oc) * 100 + ic) * 5 + ky) * 5 + kx];
    }
    o[i] = (_Float16)v;
}

// ---- w_kp f32 [441][100] -> f16 [448][128]
__global__ void cvt_wkp_k(const float* __restrict__ w, _Float16* __restrict__ o) {
    size_t i = (size_t)blockIdx.x * blockDim.x + threadIdx.x;
    if (i >= (size_t)448 * 128) return;
    int ic = i & 127;
    int oc = (int)(i >> 7);
    float v = (oc < 441 && ic < 100) ? w[(size_t)oc * 100 + ic] : 0.f;
    o[i] = (_Float16)v;
}

// ---- input conv 3->100, 5x5 SAME, relu; out layout [B][H][W][128] f16, pad zeroed
__global__ void conv_in_k(const float* __restrict__ img, const float* __restrict__ w,
                          const float* __restrict__ bias, _Float16* __restrict__ out) {
    size_t i = (size_t)blockIdx.x * blockDim.x + threadIdx.x;
    if (i >= (size_t)2 * kHW * kHW * kCPAD) return;
    int c = i & (kCPAD - 1);
    size_t t = i >> 7;
    int x = (int)(t % kHW); t /= kHW;
    int y = (int)(t % kHW);
    int b = (int)(t / kHW);
    float acc = 0.f;
    if (c < 100) {
        acc = bias[c];
        for (int ci = 0; ci < 3; ++ci)
            for (int ky = 0; ky < 5; ++ky) {
                int yy = y + ky - 2;
                if (yy < 0 || yy >= kHW) continue;
                for (int kx = 0; kx < 5; ++kx) {
                    int xx = x + kx - 2;
                    if (xx < 0 || xx >= kHW) continue;
                    acc += w[((c * 3 + ci) * 5 + ky) * 5 + kx] *
                           img[(((size_t)b * 3 + ci) * kHW + yy) * kHW + xx];
                }
            }
        acc = fmaxf(acc, 0.f);
    }
    out[i] = (_Float16)acc;
}

// ---- hidden conv 100->100, 5x5 SAME, relu — implicit GEMM via WMMA f16
// block = 128 threads (4 waves); block tile = 128 outC x 32 pixels
// each wave: 32 outC x 32 pixels = 4 accumulators (independent WMMA chains)
// grid.x = (W/32) * H * B
__global__ void __launch_bounds__(128) conv_hidden_k(
        const _Float16* __restrict__ actIn, const _Float16* __restrict__ wl,
        const float* __restrict__ bias, _Float16* __restrict__ actOut) {
    __shared__ _Float16 sAct[5][36][kCSTR];   // rows y0-2..y0+2, cols x0-2..x0+33

    int bx = blockIdx.x;
    int x0 = (bx % 6) * 32;
    int y0 = (bx / 6) % kHW;
    int b  = bx / (6 * kHW);
    int tid = threadIdx.x;

    // stage halo tile: 5*36 pixel-vectors of 128 f16 (16 x 16B chunks each)
    // in-range: async copy straight into LDS (ASYNCcnt); pad: ds-store zeros
    for (int i = tid; i < 5 * 36 * 16; i += 128) {
        int q   = i & 15;
        int col = (i >> 4) % 36;
        int r   = i / (36 * 16);
        int gy = y0 + r - 2, gx = x0 + col - 2;
        unsigned ldsOff = (unsigned)((((r * 36 + col) * kCSTR) + q * 8) * 2);
        if (gy >= 0 && gy < kHW && gx >= 0 && gx < kHW) {
            const _Float16* gp =
                actIn + (((size_t)b * kHW + gy) * kHW + gx) * kCPAD + q * 8;
            asm volatile("global_load_async_to_lds_b128 %0, %1, off"
                         :: "v"(ldsOff), "v"((unsigned long long)(uintptr_t)gp)
                         : "memory");
        } else {
            v8h z = {};
            *(v8h*)(&sAct[r][col][q * 8]) = z;
        }
    }
    asm volatile("s_wait_asynccnt 0" ::: "memory");
    __syncthreads();

    int wv        = tid >> 5;
    unsigned lane = tid & 31u;
    int selA = (lane >> 4) & 1;
    int lpix = lane & 15;
    int ocT  = wv * 32;                 // this wave's 32 outC
    int M0   = ocT + lpix;              // A tile0 row; tile1 = +16

    v8f acc00 = {}, acc01 = {}, acc10 = {}, acc11 = {};
    for (int tap = 0; tap < 25; ++tap) {
        int dy = tap / 5, dx = tap % 5;
        const _Float16* wrow0 = wl + ((size_t)tap * 128 + M0) * 128;
        const _Float16* wrow1 = wrow0 + 16 * 128;
        const _Float16* brow0 = &sAct[dy][lpix + dx][0];
        const _Float16* brow1 = brow0 + 16 * kCSTR;
        if (tap < 24) {                 // prefetch next tap's weight rows (L2 stream)
            __builtin_prefetch(wrow0 + 128 * 128, 0, 0);
            __builtin_prefetch(wrow1 + 128 * 128, 0, 0);
        }
        for (int kc = 0; kc < 128; kc += 32) {
            v16h a0 = ldA(wrow0 + kc + selA * 16, wrow0 + kc + selA * 16 + 8);
            v16h a1 = ldA(wrow1 + kc + selA * 16, wrow1 + kc + selA * 16 + 8);
            v16h b0 = ldB(brow0 + kc + selA * 16);
            v16h b1 = ldB(brow1 + kc + selA * 16);
            acc00 = wmma16(a0, b0, acc00);
            acc01 = wmma16(a0, b1, acc01);
            acc10 = wmma16(a1, b0, acc10);
            acc11 = wmma16(a1, b1, acc11);
        }
    }

    size_t rowBase = ((size_t)b * kHW + y0) * kHW;
    {
        v8f accs[4] = {acc00, acc01, acc10, acc11};
        for (int t = 0; t < 4; ++t) {
            int ocBase  = ocT + (t >> 1) * 16;
            int pixBase = (t & 1) * 16;
            size_t obase = (rowBase + x0 + pixBase + lpix) * kCPAD;
            for (int r = 0; r < 8; ++r) {
                int oc = ocBase + r + 8 * selA;
                float v = 0.f;
                if (oc < 100) v = fmaxf(accs[t][r] + bias[oc], 0.f);
                actOut[obase + oc] = (_Float16)v;
            }
        }
    }
}

// ---- fused 1x1 conv (100->441) + softmax + 21x21 kernel application
// block = 128 threads (4 waves) per 16-pixel group; grid = (W/16)*H*B
__global__ void __launch_bounds__(128) kp_apply_k(
        const _Float16* __restrict__ feat, const _Float16* __restrict__ wkp,
        const float* __restrict__ bkp, const float* __restrict__ img,
        float* __restrict__ out) {
    __shared__ _Float16 sFeat[16][kCSTR];
    __shared__ float    sLog[16][448];

    int gid = blockIdx.x;
    int x0 = (gid % 12) * 16;
    int y  = (gid / 12) % kHW;
    int b  = gid / (12 * kHW);
    int tid = threadIdx.x;

    for (int i = tid; i < 16 * 16; i += 128) {
        int q = i & 15, pix = i >> 4;
        *(v8h*)(&sFeat[pix][q * 8]) =
            *(const v8h*)(feat + (((size_t)b * kHW + y) * kHW + x0 + pix) * kCPAD + q * 8);
    }
    __syncthreads();

    int wv = tid >> 5;
    unsigned lane = tid & 31u;
    int selA = (lane >> 4) & 1;
    const _Float16* brow = &sFeat[lane & 15][0];

    for (int t = 0; t < 7; ++t) {                 // 4 waves x 7 tiles = 448 channels
        int ocTile = (wv * 7 + t) * 16;
        int M = ocTile + (lane & 15);
        const _Float16* wrow = wkp + (size_t)M * 128;
        v8f acc = {};
        for (int kc = 0; kc < 128; kc += 32) {
            v16h a  = ldA(wrow + kc + selA * 16, wrow + kc + selA * 16 + 8);
            v16h bf = ldB(brow + kc + selA * 16);
            acc = wmma16(a, bf, acc);
        }
        for (int r = 0; r < 8; ++r) {
            int oc = ocTile + r + 8 * selA;
            sLog[lane & 15][oc] = (oc < 441) ? (acc[r] + bkp[oc]) : -1e30f;
        }
    }
    __syncthreads();

    if (tid < 16) {                               // per-pixel softmax over 441
        float mx = -1e30f;
        for (int p = 0; p < 441; ++p) mx = fmaxf(mx, sLog[tid][p]);
        float s = 0.f;
        for (int p = 0; p < 441; ++p) {
            float e = __expf(sLog[tid][p] - mx);
            sLog[tid][p] = e; s += e;
        }
        float inv = 1.f / s;
        for (int p = 0; p < 441; ++p) sLog[tid][p] *= inv;
    }
    __syncthreads();

    if (tid < 48) {                               // 16 pixels x 3 channels
        int pix = tid & 15, c = tid >> 4;
        int x = x0 + pix;
        float s = 0.f;
        for (int p = 0; p < 441; ++p) {
            int yy = y + p / 21 - 10, xx = x + p % 21 - 10;
            if (yy >= 0 && yy < kHW && xx >= 0 && xx < kHW)
                s += sLog[pix][p] * img[(((size_t)b * 3 + c) * kHW + yy) * kHW + xx];
        }
        out[(((size_t)b * 3 + c) * kHW + y) * kHW + x] = s;
    }
}

extern "C" void kernel_launch(void* const* d_in, const int* in_sizes, int n_in,
                              void* d_out, int out_size, void* d_ws, size_t ws_size,
                              hipStream_t stream) {
    const float* noisy    = (const float*)d_in[0];
    const float* w_in     = (const float*)d_in[1];
    const float* b_in     = (const float*)d_in[2];
    const float* w_hidden = (const float*)d_in[3];
    const float* b_hidden = (const float*)d_in[4];
    const float* w_kp     = (const float*)d_in[5];
    const float* b_kp     = (const float*)d_in[6];
    float* outp = (float*)d_out;

    const size_t actElems = (size_t)2 * kHW * kHW * kCPAD;     // 9,437,184 halves
    _Float16* actA = (_Float16*)d_ws;
    _Float16* actB = actA + actElems;
    _Float16* whid = actB + actElems;                          // [7][25][128][128]
    _Float16* wkpq = whid + (size_t)7 * 25 * 128 * 128;        // [448][128]

    {   size_t n = (size_t)7 * 25 * 128 * 128;
        cvt_whid_k<<<dim3((unsigned)((n + 255) / 256)), dim3(256), 0, stream>>>(w_hidden, whid); }
    {   size_t n = (size_t)448 * 128;
        cvt_wkp_k<<<dim3((unsigned)((n + 255) / 256)), dim3(256), 0, stream>>>(w_kp, wkpq); }
    {   size_t n = actElems;
        conv_in_k<<<dim3((unsigned)((n + 255) / 256)), dim3(256), 0, stream>>>(noisy, w_in, b_in, actA); }

    for (int l = 0; l < 7; ++l) {
        const _Float16* in = (l & 1) ? actB : actA;
        _Float16*       o  = (l & 1) ? actA : actB;
        conv_hidden_k<<<dim3(6 * kHW * 2), dim3(128), 0, stream>>>(
            in, whid + (size_t)l * 25 * 128 * 128, b_hidden + l * 100, o);
    }
    // after 7 layers (l=0..6), final activations are in actB
    kp_apply_k<<<dim3(12 * kHW * 2), dim3(128), 0, stream>>>(actB, wkpq, b_kp, noisy, outp);
}
